// LSTM_4234837754277
// MI455X (gfx1250) — 16-bit and 8-bit matrix operand order repaired and checked by reading, compiled, not run on hardware
//
#include <hip/hip_runtime.h>
#include <hip/hip_bf16.h>
#include <math.h>


typedef __attribute__((ext_vector_type(8)))  int    v8i;
typedef __attribute__((ext_vector_type(8)))  float  v8f;
typedef __attribute__((ext_vector_type(16))) __bf16 v16bf;

#define Tlen 512
#define Bsz  128
#define INsz 4
#define Hd   256
#define Gd   1024
#define WSTR 264

__device__ __forceinline__ unsigned f32_to_e4m3(float f) {
  unsigned u = __float_as_uint(f);
  unsigned s = (u >> 24) & 0x80u;
  float a = fabsf(f);
  if (!(a >= 0.001953125f)) return s;
  if (a > 448.f) a = 448.f;
  unsigned ua = __float_as_uint(a);
  int e = (int)((ua >> 23) & 0xFF) - 127;
  if (e < -6) {
    int m = (int)(a * 512.0f + 0.5f);
    if (m >= 8) return s | 0x08u;
    return s | (unsigned)m;
  }
  unsigned r = ua + 0x7FFFFu + ((ua >> 20) & 1u);
  int e2 = (int)((r >> 23) & 0xFF) - 127;
  unsigned m3 = (r >> 20) & 7u;
  int ef = e2 + 7;
  if (ef >= 16 || (ef == 15 && m3 == 7u)) { ef = 15; m3 = 6u; }
  return s | ((unsigned)ef << 3) | m3;
}

__device__ __forceinline__ unsigned short f32_to_bf16_bits(float f) {
  unsigned u = __float_as_uint(f);
  u += 0x7FFFu + ((u >> 16) & 1u);
  return (unsigned short)(u >> 16);
}

__device__ __forceinline__ float sigm(float v) { return 1.0f / (1.0f + __expf(-v)); }

template <int LAYER>
__global__ __launch_bounds__(256)
void lstm_scan(const float* __restrict__ x,
               const __hip_bfloat16* __restrict__ y0,
               const float* __restrict__ w_ih0,
               const __hip_bfloat16* __restrict__ w1b,
               const float* __restrict__ w_hh,
               const float* __restrict__ bias,
               __hip_bfloat16* __restrict__ y_out,
               float* __restrict__ y1last)
{
  __shared__ unsigned char h8[16 * WSTR];

  const int dir = blockIdx.x >> 3;
  const int r0  = (blockIdx.x & 7) * 16;
  const int tid = threadIdx.x;
  const int lane = tid & 31;
  const int wv   = tid >> 5;
  const int ln   = lane & 15;
  const int hi   = lane >> 4;
  const int jb   = wv * 32;

  const float* Wd = w_hh + (size_t)dir * Gd * Hd;
  v8i breg[4][8];
  #pragma unroll
  for (int kt = 0; kt < 4; ++kt)
    #pragma unroll
    for (int g = 0; g < 4; ++g)
      #pragma unroll
      for (int s = 0; s < 2; ++s) {
        const int col = g * Hd + jb + s * 16 + ln;
        union { v8i v; unsigned u[8]; } Bm;
        #pragma unroll
        for (int v = 0; v < 8; ++v) {
          const int k = kt * 64 + (v & 3) * 4 + (v >> 2) * 32 + hi * 16;
          const float4 w4 = ((const float4*)(Wd + (size_t)col * Hd + k))[0];
          Bm.u[v] = f32_to_e4m3(w4.x) | (f32_to_e4m3(w4.y) << 8) |
                    (f32_to_e4m3(w4.z) << 16) | (f32_to_e4m3(w4.w) << 24);
        }
        breg[kt][g * 2 + s] = Bm.v;
      }

  for (int i = tid; i < 16 * WSTR; i += 256) h8[i] = 0;

  float  bb[4][2];
  float4 wx[4][2];
  #pragma unroll
  for (int g = 0; g < 4; ++g)
    #pragma unroll
    for (int s = 0; s < 2; ++s) {
      const int col = g * Hd + jb + s * 16 + ln;
      bb[g][s] = bias[dir * Gd + col];
      if (LAYER == 0) wx[g][s] = ((const float4*)(w_ih0 + ((size_t)dir * Gd + col) * INsz))[0];
      else            wx[g][s] = make_float4(0.f, 0.f, 0.f, 0.f);
    }

  float c[2][8];
  #pragma unroll
  for (int s = 0; s < 2; ++s)
    #pragma unroll
    for (int r = 0; r < 8; ++r) c[s][r] = 0.f;

  v8f zero;
  #pragma unroll
  for (int i = 0; i < 8; ++i) zero[i] = 0.f;

  __syncthreads();

  for (int step = 0; step < Tlen; ++step) {
    const int t = dir ? (Tlen - 1 - step) : step;

    v8f acc[8];
    #pragma unroll
    for (int gs = 0; gs < 8; ++gs) acc[gs] = zero;

    #pragma unroll
    for (int kt = 0; kt < 4; ++kt) {
      union { v8i v; unsigned u[8]; } A;
      #pragma unroll
      for (int v = 0; v < 8; ++v) {
        const int k = kt * 64 + (v & 3) * 4 + (v >> 2) * 32 + hi * 16;
        A.u[v] = *(const unsigned*)(h8 + ln * WSTR + k);
      }
      #pragma unroll
      for (int gs = 0; gs < 8; ++gs)
        acc[gs] = __builtin_amdgcn_wmma_f32_16x16x64_fp8_fp8(
            A.v, breg[kt][gs], (short)0, acc[gs], false, false);
    }

    if (LAYER == 1) {
      const unsigned short* yrow =
          (const unsigned short*)y0 + ((size_t)t * Bsz + (r0 + ln)) * (2 * Hd);
      #pragma unroll 4
      for (int kt = 0; kt < 16; ++kt) {
        union { v16bf v; unsigned u[8]; } A;
        #pragma unroll
        for (int v = 0; v < 8; ++v) {
          const int k = kt * 32 + (v & 3) * 2 + (v >> 2) * 16 + hi * 8;
          A.u[v] = *(const unsigned*)(yrow + k);
        }
        #pragma unroll
        for (int g = 0; g < 4; ++g)
          #pragma unroll
          for (int s = 0; s < 2; ++s) {
            const int col = g * Hd + jb + s * 16 + ln;
            const unsigned short* wrow =
                (const unsigned short*)w1b + ((size_t)dir * Gd + col) * (2 * Hd);
            union { v16bf v; unsigned u[8]; } Bm;
            #pragma unroll
            for (int v = 0; v < 8; ++v) {
              const int k = kt * 32 + (v & 3) * 2 + (v >> 2) * 16 + hi * 8;
              Bm.u[v] = *(const unsigned*)(wrow + k);
            }
            acc[g * 2 + s] = __builtin_amdgcn_wmma_f32_16x16x32_bf16(
                false, A.v, false, Bm.v, (short)0, acc[g * 2 + s], false, false);
          }
      }
      const int tn = dir ? (t - 1) : (t + 1);
      if (tn >= 0 && tn < Tlen)
        __builtin_prefetch((const void*)((const unsigned short*)y0 +
                           ((size_t)tn * Bsz + (r0 + ln)) * (2 * Hd)), 0, 1);
    }

    __syncthreads();

    #pragma unroll
    for (int s = 0; s < 2; ++s) {
      const int j = jb + s * 16 + ln;
      #pragma unroll
      for (int r = 0; r < 8; ++r) {
        const int row  = r + hi * 8;
        const int brow = r0 + row;
        float gi = acc[0 * 2 + s][r] + bb[0][s];
        float gf = acc[1 * 2 + s][r] + bb[1][s];
        float gg = acc[2 * 2 + s][r] + bb[2][s];
        float go = acc[3 * 2 + s][r] + bb[3][s];
        if (LAYER == 0) {
          const float4 xv = ((const float4*)(x + ((size_t)brow * Tlen + t) * INsz))[0];
          gi += xv.x * wx[0][s].x + xv.y * wx[0][s].y + xv.z * wx[0][s].z + xv.w * wx[0][s].w;
          gf += xv.x * wx[1][s].x + xv.y * wx[1][s].y + xv.z * wx[1][s].z + xv.w * wx[1][s].w;
          gg += xv.x * wx[2][s].x + xv.y * wx[2][s].y + xv.z * wx[2][s].z + xv.w * wx[2][s].w;
          go += xv.x * wx[3][s].x + xv.y * wx[3][s].y + xv.z * wx[3][s].z + xv.w * wx[3][s].w;
        }
        const float iv = sigm(gi), fv = sigm(gf);
        const float gv = tanhf(gg), ov = sigm(go);
        const float cn = fv * c[s][r] + iv * gv;
        c[s][r] = cn;
        const float h = ov * tanhf(cn);
        h8[row * WSTR + j] = (unsigned char)f32_to_e4m3(h);
        if (LAYER == 0) {
          ((unsigned short*)y_out)[((size_t)t * Bsz + brow) * (2 * Hd) + dir * Hd + j] =
              f32_to_bf16_bits(h);
        } else if (t == Tlen - 1) {
          y1last[(size_t)brow * (2 * Hd) + dir * Hd + j] = h;
        }
      }
    }
    __syncthreads();
  }
}

__global__ void cvt_bf16_kernel(const float* __restrict__ src,
                                unsigned short* __restrict__ dst, int n) {
  int i = blockIdx.x * blockDim.x + threadIdx.x;
  if (i < n) dst[i] = f32_to_bf16_bits(src[i]);
}

__global__ void head_kernel(const float* __restrict__ y1last,
                            const float* __restrict__ w_out,
                            const float* __restrict__ b_out,
                            float* __restrict__ out) {
  int b = blockIdx.x * blockDim.x + threadIdx.x;
  if (b >= Bsz) return;
  const float* v = y1last + (size_t)b * (2 * Hd);
  float lg[3];
  #pragma unroll
  for (int cc = 0; cc < 3; ++cc) {
    float a = b_out[cc];
    const float* w = w_out + cc * Hd;
    for (int j = 0; j < Hd; ++j) a += (v[j] + v[Hd + j]) * w[j];
    lg[cc] = a;
  }
  float m = fmaxf(lg[0], fmaxf(lg[1], lg[2]));
  float e0 = __expf(lg[0] - m), e1 = __expf(lg[1] - m), e2 = __expf(lg[2] - m);
  float inv = 1.0f / (e0 + e1 + e2);
  out[b * 3 + 0] = e0 * inv;
  out[b * 3 + 1] = e1 * inv;
  out[b * 3 + 2] = e2 * inv;
}

extern "C" void kernel_launch(void* const* d_in, const int* in_sizes, int n_in,
                              void* d_out, int out_size, void* d_ws, size_t ws_size,
                              hipStream_t stream) {
  const float* x     = (const float*)d_in[0];
  const float* w_ih0 = (const float*)d_in[1];
  const float* w_hh0 = (const float*)d_in[2];
  const float* b0    = (const float*)d_in[3];
  const float* w_ih1 = (const float*)d_in[4];
  const float* w_hh1 = (const float*)d_in[5];
  const float* b1    = (const float*)d_in[6];
  const float* w_out = (const float*)d_in[7];
  const float* b_out = (const float*)d_in[8];

  char* ws = (char*)d_ws;
  __hip_bfloat16* y0  = (__hip_bfloat16*)ws;
  __hip_bfloat16* w1b = (__hip_bfloat16*)(ws + (size_t)Tlen * Bsz * 512 * 2);
  float* y1last = (float*)(ws + (size_t)Tlen * Bsz * 512 * 2 + (size_t)2 * Gd * 512 * 2);

  const int n1 = 2 * Gd * 512;
  cvt_bf16_kernel<<<(n1 + 255) / 256, 256, 0, stream>>>(w_ih1, (unsigned short*)w1b, n1);

  lstm_scan<0><<<16, 256, 0, stream>>>(x, nullptr, w_ih0, nullptr,
                                       w_hh0, b0, y0, nullptr);
  lstm_scan<1><<<16, 256, 0, stream>>>(nullptr, y0, nullptr, w1b,
                                       w_hh1, b1, nullptr, y1last);
  head_kernel<<<1, 128, 0, stream>>>(y1last, w_out, b_out, (float*)d_out);
}
